// RNN_LSTM_46084999086833
// MI455X (gfx1250) — matrix operand order repaired and checked by reading, not compiled, not run on hardware
//
#include <hip/hip_runtime.h>
#include <hip/hip_bf16.h>
#include <stdint.h>
#include <stddef.h>

// Problem constants (match reference)
#define B_   128
#define T_   784
#define H_   1024
#define C_   10
#define NWG  64      // = H_/16, one workgroup per 16 hidden columns
#define NTH  256     // 8 wave32s per WG
#define BH_  (B_ * H_)
#define W4H_ (4 * H_ * H_)   // 4,194,304 elements per recurrent weight

typedef __attribute__((ext_vector_type(16))) __bf16 v16bf;
typedef __attribute__((ext_vector_type(8)))  __bf16 v8bf;
typedef __attribute__((ext_vector_type(8)))  float  v8f;

// ---------------- helpers -------------------------------------------------

__device__ __forceinline__ unsigned short f2bf(float f) {
  unsigned int u = __float_as_uint(f);
  u += 0x7FFFu + ((u >> 16) & 1u);          // round-to-nearest-even
  return (unsigned short)(u >> 16);
}

// Branch-free activations on the hardware v_exp_f32 (exp2) / v_rcp_f32 path.
// sigmoid(x) = 1 / (1 + 2^(-x*log2(e)));  tanh(x) = 2*sigmoid(2x) - 1
// Saturation at +/-inf is exact (exp2->inf => rcp->0 ; exp2->0 => rcp(1)=1).
#define LOG2E_ 1.4426950408889634f
__device__ __forceinline__ float sigf(float x) {
  return __builtin_amdgcn_rcpf(1.0f + __builtin_amdgcn_exp2f(-LOG2E_ * x));
}
__device__ __forceinline__ float tanh_fast(float x) {
  return 2.0f * __builtin_amdgcn_rcpf(1.0f + __builtin_amdgcn_exp2f(-2.0f * LOG2E_ * x)) - 1.0f;
}

// A-fragment (16x32 bf16, M=batch rows, K contiguous in memory, row-major [M][K])
// lanes 0-15 : row M=lane,    K = k0..k0+7  and k0+16..k0+23
// lanes 16-31: row M=lane-16, K = k0+8..+15 and k0+24..+31
__device__ __forceinline__ v16bf load_a_frag(const unsigned short* A, int lda,
                                             int mbase, int k0, int lane) {
  const unsigned short* p =
      A + (size_t)(mbase + (lane & 15)) * lda + k0 + ((lane >> 4) << 3);
  v8bf lo = *(const v8bf*)p;
  v8bf hi = *(const v8bf*)(p + 16);
  return __builtin_shufflevector(lo, hi, 0,1,2,3,4,5,6,7,8,9,10,11,12,13,14,15);
}

// B-fragment (32x16 bf16, N = weight rows [4H][H], K contiguous)
// lanes 0-15 : col N=lane, K = k0..k0+7 and k0+16..k0+23 ; lanes 16-31: K = k0+8..+15 and k0+24..+31
__device__ __forceinline__ v16bf load_b_frag(const unsigned short* W, int ldb,
                                             int nbase, int k0, int lane) {
  const unsigned short* p =
      W + (size_t)(nbase + (lane & 15)) * ldb + k0 + ((lane >> 4) << 3);
  v8bf lo = *(const v8bf*)p;
  v8bf hi = *(const v8bf*)(p + 16);
  return __builtin_shufflevector(lo, hi, 0,1,2,3,4,5,6,7,8,9,10,11,12,13,14,15);
}

__device__ __forceinline__ v8f wmma_bf16(v16bf a, v16bf b, v8f c) {
  return __builtin_amdgcn_wmma_f32_16x16x32_bf16(false, a, false, b,
                                                 (short)0, c, false, false);
}

// Grid-wide barrier (sense via generation counter). All WGs resident (NWG=64).
__device__ __forceinline__ void grid_sync(unsigned int* cnt, unsigned int* gen) {
  __threadfence();
  __syncthreads();
  if (threadIdx.x == 0) {
    unsigned int g = atomicAdd(gen, 0u);
    if (atomicAdd(cnt, 1u) == (unsigned)NWG - 1u) {
      atomicExch(cnt, 0u);
      __threadfence();
      atomicAdd(gen, 1u);
    } else {
      while (atomicAdd(gen, 0u) == g) { __builtin_amdgcn_s_sleep(2); }
    }
  }
  __syncthreads();
  __threadfence();
}

// ---------------- prologue kernels ---------------------------------------

__global__ void k_zero32(unsigned int* p, size_t n) {
  size_t i = (size_t)blockIdx.x * blockDim.x + threadIdx.x;
  size_t stride = (size_t)gridDim.x * blockDim.x;
  for (; i < n; i += stride) p[i] = 0u;
}

__global__ void k_cvt_bf16(const float* __restrict__ src,
                           unsigned short* __restrict__ dst, int n) {
  int i = blockIdx.x * blockDim.x + threadIdx.x;
  if (i < n) dst[i] = f2bf(src[i]);
}

// ---------------- persistent LSTM kernel ----------------------------------

__global__ __launch_bounds__(NTH, 1) void lstm_persistent(
    const float* __restrict__ x,      // [B, T]
    const float* __restrict__ wih1,   // [4H] (input size 1)
    const float* __restrict__ bih1, const float* __restrict__ bhh1,
    const float* __restrict__ bih2, const float* __restrict__ bhh2,
    const float* __restrict__ fcw,  const float* __restrict__ fcb,
    const unsigned short* __restrict__ whh1,  // bf16 [4H, H]
    const unsigned short* __restrict__ wih2,  // bf16 [4H, H]
    const unsigned short* __restrict__ whh2,  // bf16 [4H, H]
    unsigned short* __restrict__ h1bf,        // bf16, double buffered [2][B,H]
    unsigned short* __restrict__ h2bf,        // bf16, double buffered [2][B,H]
    unsigned short* __restrict__ c1bf,        // bf16 [B,H]
    float* __restrict__ c1f, float* __restrict__ c2f,  // f32 [B,H]
    unsigned int* __restrict__ bar,
    float* __restrict__ out)          // [B, T, C]
{
  const int lane = threadIdx.x & 31;
  const int w    = threadIdx.x >> 5;            // wave id in WG [0,8)
  const int jt   = blockIdx.x;                  // hidden-column tile
  const int jcol = jt * 16;
  const int j    = jcol + (lane & 15);          // this lane's hidden column
  const int mb   = w * 16;                      // batch-row base for this wave
  const int mtop = (lane >> 4) << 3;            // 0 or 8 (C/D layout M offset)
  const int gw   = blockIdx.x * 8 + w;          // global wave id [0,512)
  const v8f vz   = {0.f,0.f,0.f,0.f,0.f,0.f,0.f,0.f};

  // Per-column bias terms (uniform over batch)
  float bs1[4], wi1[4], bs2[4];
  #pragma unroll
  for (int s = 0; s < 4; ++s) {
    bs1[s] = bih1[s*H_ + j] + bhh1[s*H_ + j];
    wi1[s] = wih1[s*H_ + j];
    bs2[s] = bih2[s*H_ + j] + bhh2[s*H_ + j];
  }

  #pragma unroll 1
  for (int t = 0; t < T_; ++t) {
    const unsigned short* h1r = h1bf + (size_t)(t & 1) * BH_;
    unsigned short*       h1w = h1bf + (size_t)((t + 1) & 1) * BH_;
    const unsigned short* h2r = h2bf + (size_t)(t & 1) * BH_;
    unsigned short*       h2w = h2bf + (size_t)((t + 1) & 1) * BH_;

    // ---- phase 1: layer-1 gates = h1 @ W_hh1^T (+ x*W_ih1 + biases) ------
    v8f acc[4] = {vz, vz, vz, vz};
    for (int k0 = 0; k0 < H_; k0 += 32) {
      v16bf a = load_a_frag(h1r, H_, mb, k0, lane);
      #pragma unroll
      for (int s = 0; s < 4; ++s) {
        v16bf bf = load_b_frag(whh1, H_, s*H_ + jcol, k0, lane);
        acc[s] = wmma_bf16(a, bf, acc[s]);
      }
    }
    #pragma unroll
    for (int i = 0; i < 8; ++i) {
      int b = mb + mtop + i;                 // C/D layout: lanes<16 M=i, else M=8+i
      float xv = x[(size_t)b * T_ + t];
      float ig = acc[0][i] + bs1[0] + xv * wi1[0];
      float fg = acc[1][i] + bs1[1] + xv * wi1[1];
      float gg = acc[2][i] + bs1[2] + xv * wi1[2];
      float og = acc[3][i] + bs1[3] + xv * wi1[3];
      size_t idx = (size_t)b * H_ + j;
      float cp = c1f[idx];
      float cn = sigf(fg) * cp + sigf(ig) * tanh_fast(gg);
      float hn = sigf(og) * tanh_fast(cn);
      c1f[idx]  = cn;
      c1bf[idx] = f2bf(cn);
      h1w[idx]  = f2bf(hn);
    }
    // Folded-in FC for step t-1 (reads c2f written in phase 2 of t-1; c2f is
    // next overwritten only in phase 2 of t, i.e. after the coming barrier).
    if (t > 0) {
      for (int task = gw; task < B_ * C_; task += NWG * 8) {
        int b = task / C_, c = task % C_;
        float s = 0.f;
        for (int k = lane; k < H_; k += 32)
          s += c2f[(size_t)b * H_ + k] * fcw[(size_t)c * H_ + k];
        #pragma unroll
        for (int off = 16; off > 0; off >>= 1) s += __shfl_xor(s, off, 32);
        if (lane == 0) out[((size_t)b * T_ + (t - 1)) * C_ + c] = s + fcb[c];
      }
    }
    grid_sync(bar, bar + 1);

    // ---- phase 2: layer-2 gates = c1 @ W_ih2^T + h2 @ W_hh2^T ------------
    acc[0] = vz; acc[1] = vz; acc[2] = vz; acc[3] = vz;
    for (int k0 = 0; k0 < H_; k0 += 32) {
      v16bf a = load_a_frag(c1bf, H_, mb, k0, lane);
      #pragma unroll
      for (int s = 0; s < 4; ++s) {
        v16bf bf = load_b_frag(wih2, H_, s*H_ + jcol, k0, lane);
        acc[s] = wmma_bf16(a, bf, acc[s]);
      }
    }
    for (int k0 = 0; k0 < H_; k0 += 32) {
      v16bf a = load_a_frag(h2r, H_, mb, k0, lane);
      #pragma unroll
      for (int s = 0; s < 4; ++s) {
        v16bf bf = load_b_frag(whh2, H_, s*H_ + jcol, k0, lane);
        acc[s] = wmma_bf16(a, bf, acc[s]);
      }
    }
    #pragma unroll
    for (int i = 0; i < 8; ++i) {
      int b = mb + mtop + i;
      float ig = acc[0][i] + bs2[0];
      float fg = acc[1][i] + bs2[1];
      float gg = acc[2][i] + bs2[2];
      float og = acc[3][i] + bs2[3];
      size_t idx = (size_t)b * H_ + j;
      float cp = c2f[idx];
      float cn = sigf(fg) * cp + sigf(ig) * tanh_fast(gg);
      float hn = sigf(og) * tanh_fast(cn);
      c2f[idx] = cn;
      h2w[idx] = f2bf(hn);
    }
    grid_sync(bar, bar + 1);
  }

  // Epilogue FC for the last step
  for (int task = gw; task < B_ * C_; task += NWG * 8) {
    int b = task / C_, c = task % C_;
    float s = 0.f;
    for (int k = lane; k < H_; k += 32)
      s += c2f[(size_t)b * H_ + k] * fcw[(size_t)c * H_ + k];
    #pragma unroll
    for (int off = 16; off > 0; off >>= 1) s += __shfl_xor(s, off, 32);
    if (lane == 0) out[((size_t)b * T_ + (T_ - 1)) * C_ + c] = s + fcb[c];
  }
}

// ---------------- launch ---------------------------------------------------

extern "C" void kernel_launch(void* const* d_in, const int* in_sizes, int n_in,
                              void* d_out, int out_size, void* d_ws, size_t ws_size,
                              hipStream_t stream) {
  const float* x    = (const float*)d_in[0];
  const float* Wih1 = (const float*)d_in[1];
  const float* Whh1 = (const float*)d_in[2];
  const float* bih1 = (const float*)d_in[3];
  const float* bhh1 = (const float*)d_in[4];
  const float* Wih2 = (const float*)d_in[5];
  const float* Whh2 = (const float*)d_in[6];
  const float* bih2 = (const float*)d_in[7];
  const float* bhh2 = (const float*)d_in[8];
  const float* fcw  = (const float*)d_in[9];
  const float* fcb  = (const float*)d_in[10];
  float* out = (float*)d_out;

  // Workspace carve-up (aligned to 256B)
  char* ws = (char*)d_ws;
  size_t off = 0;
  auto carve = [&](size_t bytes) -> char* {
    char* p = ws + off;
    off = (off + bytes + 255) & ~(size_t)255;
    return p;
  };
  unsigned short* whh1bf = (unsigned short*)carve((size_t)W4H_ * 2);
  unsigned short* wih2bf = (unsigned short*)carve((size_t)W4H_ * 2);
  unsigned short* whh2bf = (unsigned short*)carve((size_t)W4H_ * 2);
  char* state_base = ws + off;
  unsigned short* h1bf = (unsigned short*)carve((size_t)2 * BH_ * 2);
  unsigned short* h2bf = (unsigned short*)carve((size_t)2 * BH_ * 2);
  unsigned short* c1bf = (unsigned short*)carve((size_t)BH_ * 2);
  float*          c1f  = (float*)carve((size_t)BH_ * 4);
  float*          c2f  = (float*)carve((size_t)BH_ * 4);
  unsigned int*   bar  = (unsigned int*)carve(256);
  size_t state_words = ((ws + off) - state_base) / 4;

  // Zero recurrent state + barrier (d_ws is poisoned; must re-init each call)
  k_zero32<<<512, 256, 0, stream>>>((unsigned int*)state_base, state_words);

  // Convert recurrent weights to bf16 (kept row-major [4H][H], K contiguous)
  const int cvt_blocks = (W4H_ + 255) / 256;
  k_cvt_bf16<<<cvt_blocks, 256, 0, stream>>>(Whh1, whh1bf, W4H_);
  k_cvt_bf16<<<cvt_blocks, 256, 0, stream>>>(Wih2, wih2bf, W4H_);
  k_cvt_bf16<<<cvt_blocks, 256, 0, stream>>>(Whh2, whh2bf, W4H_);

  // Persistent timestep loop
  lstm_persistent<<<NWG, NTH, 0, stream>>>(
      x, Wih1, bih1, bhh1, bih2, bhh2, fcw, fcb,
      whh1bf, wih2bf, whh2bf,
      h1bf, h2bf, c1bf, c1f, c2f, bar, out);
}
